// LinearAttention_11845519802369
// MI455X (gfx1250) — matrix operand order repaired and checked by reading, not compiled, not run on hardware
//
#include <hip/hip_runtime.h>
#include <hip/hip_bf16.h>

// ---------- vector types ----------
typedef __attribute__((ext_vector_type(16))) __bf16       v16bf;
typedef __attribute__((ext_vector_type(8)))  float        v8f;
typedef __attribute__((ext_vector_type(4)))  __bf16       bf16x4;
typedef __attribute__((ext_vector_type(4)))  unsigned int u32x4;
typedef __attribute__((ext_vector_type(8)))  int          i32x8;
typedef __attribute__((ext_vector_type(4)))  int          i32x4;
typedef __attribute__((ext_vector_type(4)))  float        f32x4;

union FragBF16 {           // one WMMA 16-bit operand: 16 bf16 = 8 VGPRs
    u32x4 u[2];
    v16bf v;
};

// ---------- TDM: 2D tile load global -> LDS (CDNA5 ISA) ----------
// Moves tile_d0 x tile_d1 elements (bf16, data_size=1 -> 2 bytes) from a row-major
// tensor with row stride `stride_elems` into LDS at byte offset `lds_addr` (packed).
__device__ __forceinline__ void tdm_load_2d_bf16(unsigned lds_addr, const void* gaddr,
                                                 unsigned stride_elems,
                                                 unsigned tile_d0, unsigned tile_d1) {
    unsigned long long ga = (unsigned long long)(uintptr_t)gaddr;
    u32x4 g0;
    g0[0] = 1u;                                        // count=1 (valid), user mode
    g0[1] = lds_addr;                                  // D#.lds_addr (bytes)
    g0[2] = (unsigned)(ga & 0xFFFFFFFFu);              // global_addr[31:0]
    g0[3] = (unsigned)((ga >> 32) & 0x01FFFFFFu) | (2u << 30);  // addr[56:32] | type=2
    i32x8 g1;
    g1[0] = (int)(1u << 16);                           // data_size=1 (2 bytes)
    g1[1] = (int)((tile_d0 & 0xFFFFu) << 16);          // tensor_dim0[15:0] = tile_d0
    g1[2] = (int)(((tile_d0 >> 16) & 0xFFFFu) | ((tile_d1 & 0xFFFFu) << 16)); // dim0 hi | dim1 lo
    g1[3] = (int)(((tile_d1 >> 16) & 0xFFFFu) | ((tile_d0 & 0xFFFFu) << 16)); // dim1 hi | tile_dim0
    g1[4] = (int)(tile_d1 & 0xFFFFu);                  // tile_dim1 ; tile_dim2 = 0 (2D)
    g1[5] = (int)stride_elems;                         // tensor_dim0_stride[31:0]
    g1[6] = 0;                                         // stride[47:32]=0, dim1_stride lo=0
    g1[7] = 0;
    i32x4 z4 = {0, 0, 0, 0};
#if __clang_major__ >= 23
    i32x8 z8 = {0, 0, 0, 0, 0, 0, 0, 0};
    __builtin_amdgcn_tensor_load_to_lds(g0, g1, z4, z4, z8, 0);
#else
    __builtin_amdgcn_tensor_load_to_lds(g0, g1, z4, z4, 0);
#endif
}

__device__ __forceinline__ unsigned lds_offset_of(const void* p) {
    return (unsigned)(uintptr_t)p;   // LDS aperture: addr[31:0] is the LDS offset
}

// ---------- fp32 -> bf16 conversion (x) ----------
__global__ __launch_bounds__(256) void cvt_f32_bf16(const float* __restrict__ in,
                                                    __bf16* __restrict__ out, int n) {
    int i = (blockIdx.x * 256 + threadIdx.x) * 4;
    if (i + 3 < n) {
        f32x4 f = *(const f32x4*)(in + i);
        bf16x4 b;
        b[0] = (__bf16)f[0]; b[1] = (__bf16)f[1];
        b[2] = (__bf16)f[2]; b[3] = (__bf16)f[3];
        *(bf16x4*)(out + i) = b;
    }
}

// ---------- fp32 [K][N] -> bf16 transposed [N][K] (weights), LDS-tiled 64x64 ----------
__global__ __launch_bounds__(256) void cvt_transpose_f32_bf16(const float* __restrict__ in,
                                                              __bf16* __restrict__ out,
                                                              int Kd, int Nd) {
    __shared__ __bf16 t[64 * 65];
    const int tid = threadIdx.x;
    const int ntx = Nd >> 6;
    const int k0 = (blockIdx.x / ntx) * 64;
    const int n0 = (blockIdx.x % ntx) * 64;
    #pragma unroll
    for (int it = 0; it < 16; ++it) {
        int lin = tid + it * 256;
        int r = lin >> 6, c = lin & 63;                 // coalesced read along n
        t[c * 65 + r] = (__bf16)in[(size_t)(k0 + r) * Nd + n0 + c];
    }
    __syncthreads();
    #pragma unroll
    for (int it = 0; it < 16; ++it) {
        int lin = tid + it * 256;
        int c = lin >> 6, r = lin & 63;                 // coalesced write along k
        out[(size_t)(n0 + c) * Kd + k0 + r] = t[c * 65 + r];
    }
}

// ---------- generic bf16 WMMA GEMM: C = act(A[M,K] @ BT[N,K]^T + bias) ----------
// B is pre-transposed in global memory -> both tiles are straight 2D TDM loads.
// Double-buffered, software-pipelined TDM: tile t+1 DMA overlaps tile t compute.
// ACT: 0 = none, 1 = elu(z)+1.  OT: __bf16 or float.
// Block tile 128x128, 256 threads (8 waves), wave tile 64x32, K-step 32.
template <int ACT, typename OT>
__global__ __launch_bounds__(256) void gemm_bf16(const __bf16* __restrict__ A,
                                                 const __bf16* __restrict__ BtG,
                                                 const float* __restrict__ bias,
                                                 OT* __restrict__ C,
                                                 int M, int N, int K) {
    __shared__ __align__(16) __bf16 Al[2][128 * 32];  // A tiles, row-major [m][k]
    __shared__ __align__(16) __bf16 Bt[2][128 * 32];  // B tiles, [n][k]

    const int tid  = threadIdx.x;
    const int lane = tid & 31;
    const int w    = tid >> 5;
    const int m0   = blockIdx.y * 128;
    const int n0   = blockIdx.x * 128;
    const int wm   = w & 1;        // 0..1 -> 64 rows each
    const int wn   = w >> 1;       // 0..3 -> 32 cols each
    const int mrow = lane & 15;
    const int khalf = (lane >> 4) & 1;
    const bool tdm_wave = (w == 0);
    const int ntiles = K >> 5;

    // prologue: kick off tile 0
    if (tdm_wave) {
        tdm_load_2d_bf16(lds_offset_of(&Al[0][0]), &A  [(size_t)m0 * K], (unsigned)K, 32u, 128u);
        tdm_load_2d_bf16(lds_offset_of(&Bt[0][0]), &BtG[(size_t)n0 * K], (unsigned)K, 32u, 128u);
    }

    v8f acc[4][2] = {};

    for (int t = 0; t < ntiles; ++t) {
        const int cur = t & 1;
        __syncthreads();   // all reads of buf[cur^1] (iteration t-1) complete
        if (tdm_wave) {
            if (t + 1 < ntiles) {
                const int kn = (t + 1) << 5;
                tdm_load_2d_bf16(lds_offset_of(&Al[cur ^ 1][0]),
                                 &A  [(size_t)m0 * K + kn], (unsigned)K, 32u, 128u);
                tdm_load_2d_bf16(lds_offset_of(&Bt[cur ^ 1][0]),
                                 &BtG[(size_t)n0 * K + kn], (unsigned)K, 32u, 128u);
                __builtin_amdgcn_s_wait_tensorcnt(2);   // tile t done; t+1 in flight
            } else {
                __builtin_amdgcn_s_wait_tensorcnt(0);   // last tile done
            }
        }
        __syncthreads();   // tile t resident in LDS
        const __bf16* Ab = &Al[cur][0];
        const __bf16* Bb = &Bt[cur][0];

        // ---- fragments + WMMA
        FragBF16 a[4];
        #pragma unroll
        for (int mi = 0; mi < 4; ++mi) {
            int row = wm * 64 + mi * 16 + mrow;
            a[mi].u[0] = *(const u32x4*)&Ab[row * 32 + khalf * 8];        // K 0-7 / 8-15
            a[mi].u[1] = *(const u32x4*)&Ab[row * 32 + 16 + khalf * 8];   // K 16-23 / 24-31
        }
        #pragma unroll
        for (int ni = 0; ni < 2; ++ni) {
            int n = wn * 32 + ni * 16 + mrow;
            FragBF16 bfr;
            bfr.u[0] = *(const u32x4*)&Bb[n * 32 + khalf * 8];            // K 0-7 / 8-15
            bfr.u[1] = *(const u32x4*)&Bb[n * 32 + 16 + khalf * 8];       // K 16-23 / 24-31
            #pragma unroll
            for (int mi = 0; mi < 4; ++mi)
                acc[mi][ni] = __builtin_amdgcn_wmma_f32_16x16x32_bf16(
                    false, a[mi].v, false, bfr.v, (short)0, acc[mi][ni], false, false);
        }
    }

    // ---- epilogue
    #pragma unroll
    for (int mi = 0; mi < 4; ++mi) {
        #pragma unroll
        for (int ni = 0; ni < 2; ++ni) {
            int Mr0  = m0 + wm * 64 + mi * 16 + (khalf << 3);
            int Ncol = n0 + wn * 32 + ni * 16 + mrow;
            float bval = bias[Ncol];
            #pragma unroll
            for (int r = 0; r < 8; ++r) {
                float z = acc[mi][ni][r] + bval;
                if (ACT == 1) z = (z > 0.0f) ? (z + 1.0f) : __expf(z);
                C[(size_t)(Mr0 + r) * N + Ncol] = (OT)z;
            }
        }
    }
}

// ---------- kv = k^T v (per head, K over 4096 tokens) + sum_k ----------
// One block per (b,h). kv stored TRANSPOSED (kvT[e][d]) in bf16 for the next stage.
__global__ __launch_bounds__(256) void kv_sumk_kernel(const __bf16* __restrict__ k,
                                                      const __bf16* __restrict__ v,
                                                      __bf16* __restrict__ kvT,
                                                      float* __restrict__ sumk) {
    __shared__ __align__(16) __bf16 kT[64 * 64];  // [d][token]
    __shared__ __align__(16) __bf16 vT[64 * 64];  // [e][token]
    __shared__ float red[256];

    const int bh = blockIdx.x;
    const int b = bh >> 4, h = bh & 15;
    const size_t rowbase = (size_t)b * 4096;
    const int col0 = h * 64;
    const int tid = threadIdx.x, lane = tid & 31, w = tid >> 5;
    const int wm = w >> 1;        // 0..3 -> M (dim) subtile
    const int wn = w & 1;         // 0..1 -> N (e) half
    const int mrow = lane & 15, khalf = (lane >> 4) & 1;
    const int dloc = tid & 63, nsub = tid >> 6;

    v8f acc[2] = {};
    float saccum = 0.0f;

    for (int nt = 0; nt < 4096; nt += 64) {
        __syncthreads();
        // stage 64 tokens x 64 dims of k and v, transposed into LDS
        #pragma unroll
        for (int it = 0; it < 4; ++it) {
            int lin = tid + it * 256;          // 1024 chunks of 4 bf16
            int n = lin >> 4, d0 = (lin & 15) * 4;
            size_t g = (rowbase + nt + n) * 1024 + col0 + d0;
            bf16x4 kk4 = *(const bf16x4*)&k[g];
            bf16x4 vv4 = *(const bf16x4*)&v[g];
            #pragma unroll
            for (int j = 0; j < 4; ++j) {
                kT[(d0 + j) * 64 + n] = kk4[j];
                vT[(d0 + j) * 64 + n] = vv4[j];
            }
        }
        __syncthreads();
        // sum_k partial from staged tile
        #pragma unroll
        for (int j = 0; j < 16; ++j) saccum += (float)kT[dloc * 64 + nsub * 16 + j];
        // WMMA: A = kT (M=dim, K=token), B = vT (N=e, K=token)
        #pragma unroll
        for (int ks = 0; ks < 2; ++ks) {
            FragBF16 a;
            int row = wm * 16 + mrow;
            a.u[0] = *(const u32x4*)&kT[row * 64 + ks * 32 + khalf * 8];
            a.u[1] = *(const u32x4*)&kT[row * 64 + ks * 32 + 16 + khalf * 8];
            #pragma unroll
            for (int ni = 0; ni < 2; ++ni) {
                int e = wn * 32 + ni * 16 + mrow;
                FragBF16 bb;
                bb.u[0] = *(const u32x4*)&vT[e * 64 + ks * 32 + khalf * 8];
                bb.u[1] = *(const u32x4*)&vT[e * 64 + ks * 32 + 16 + khalf * 8];
                acc[ni] = __builtin_amdgcn_wmma_f32_16x16x32_bf16(
                    false, a.v, false, bb.v, (short)0, acc[ni], false, false);
            }
        }
    }

    // store kvT[e][d] (bf16), per head 64x64
    __bf16* dst = kvT + (size_t)bh * 64 * 64;
    #pragma unroll
    for (int ni = 0; ni < 2; ++ni) {
        int e  = wn * 32 + ni * 16 + mrow;
        int d0 = wm * 16 + (khalf << 3);
        #pragma unroll
        for (int r = 0; r < 8; ++r) dst[e * 64 + d0 + r] = (__bf16)acc[ni][r];
    }
    // reduce sum_k
    red[tid] = saccum;
    __syncthreads();
    if (tid < 64) {
        float s = red[tid] + red[tid + 64] + red[tid + 128] + red[tid + 192];
        sumk[bh * 64 + tid] = s;
    }
}

// ---------- numerator = q @ kv ; y = numerator / (q . sum_k) ----------
// grid: (32 token-tiles of 128, 64 heads). y written bf16, heads merged.
__global__ __launch_bounds__(256) void attn_num_kernel(const __bf16* __restrict__ q,
                                                       const __bf16* __restrict__ kvT,
                                                       const float* __restrict__ sumk,
                                                       __bf16* __restrict__ y) {
    __shared__ __align__(16) __bf16 qt[128 * 64];   // [token][d] row-major
    __shared__ __align__(16) __bf16 kvl[64 * 64];   // kvT: [e][d]
    __shared__ float sk[64];
    __shared__ float den[128];

    const int tile = blockIdx.x;  // 0..31
    const int bh   = blockIdx.y;  // 0..63
    const int b = bh >> 4, h = bh & 15;
    const size_t row0 = (size_t)b * 4096 + (size_t)tile * 128;
    const int col0 = h * 64;
    const int tid = threadIdx.x, lane = tid & 31, w = tid >> 5;
    const int mrow = lane & 15, khalf = (lane >> 4) & 1;

    // q tile via TDM: 128 rows x 64 cols, row stride 1024
    if (w == 0)
        tdm_load_2d_bf16(lds_offset_of(&qt[0]), &q[row0 * 1024 + col0], 1024u, 64u, 128u);

    #pragma unroll
    for (int it = 0; it < 2; ++it) {
        int lin = tid + it * 256;          // 512 chunks of 8 bf16
        *(u32x4*)&kvl[lin * 8] = *(const u32x4*)&kvT[(size_t)bh * 4096 + lin * 8];
    }
    if (tid < 64) sk[tid] = sumk[bh * 64 + tid];
    if (w == 0) __builtin_amdgcn_s_wait_tensorcnt(0);
    __syncthreads();

    if (tid < 128) {
        float a = 0.0f;
        #pragma unroll
        for (int dd = 0; dd < 64; ++dd) a += (float)qt[tid * 64 + dd] * sk[dd];
        den[tid] = a;
    }
    __syncthreads();

    v8f acc[4] = {};
    #pragma unroll
    for (int ks = 0; ks < 2; ++ks) {
        FragBF16 a;
        int row = w * 16 + mrow;
        a.u[0] = *(const u32x4*)&qt[row * 64 + ks * 32 + khalf * 8];
        a.u[1] = *(const u32x4*)&qt[row * 64 + ks * 32 + 16 + khalf * 8];
        #pragma unroll
        for (int ni = 0; ni < 4; ++ni) {
            int e = ni * 16 + mrow;
            FragBF16 bb;
            bb.u[0] = *(const u32x4*)&kvl[e * 64 + ks * 32 + khalf * 8];
            bb.u[1] = *(const u32x4*)&kvl[e * 64 + ks * 32 + 16 + khalf * 8];
            acc[ni] = __builtin_amdgcn_wmma_f32_16x16x32_bf16(
                false, a.v, false, bb.v, (short)0, acc[ni], false, false);
        }
    }

    int t0 = w * 16 + (khalf << 3);
    #pragma unroll
    for (int ni = 0; ni < 4; ++ni) {
        int e = ni * 16 + mrow;
        #pragma unroll
        for (int r = 0; r < 8; ++r) {
            int token = t0 + r;
            float val = acc[ni][r] / den[token];
            y[(row0 + token) * 1024 + col0 + e] = (__bf16)val;
        }
    }
}

// ---------- host launch ----------
extern "C" void kernel_launch(void* const* d_in, const int* in_sizes, int n_in,
                              void* d_out, int out_size, void* d_ws, size_t ws_size,
                              hipStream_t stream) {
    (void)in_sizes; (void)n_in; (void)out_size; (void)ws_size;
    const float* x  = (const float*)d_in[0];
    const float* Wq = (const float*)d_in[1];
    const float* bq = (const float*)d_in[2];
    const float* Wk = (const float*)d_in[3];
    const float* bk = (const float*)d_in[4];
    const float* Wv = (const float*)d_in[5];
    const float* bv = (const float*)d_in[6];
    const float* Wp = (const float*)d_in[7];
    const float* bp = (const float*)d_in[8];
    float* out = (float*)d_out;

    const size_t MT = 16384, Dm = 1024;
    char* ws = (char*)d_ws;
    size_t off = 0;
    auto alloc = [&](size_t bytes) -> char* {
        char* p = ws + off;
        off += (bytes + 255) & ~(size_t)255;
        return p;
    };
    __bf16* xb   = (__bf16*)alloc(MT * Dm * 2);
    __bf16* WqT  = (__bf16*)alloc(Dm * Dm * 2);   // [N][K] transposed bf16
    __bf16* WkT  = (__bf16*)alloc(Dm * Dm * 2);
    __bf16* WvT  = (__bf16*)alloc(Dm * Dm * 2);
    __bf16* WpT  = (__bf16*)alloc(Dm * Dm * 2);
    __bf16* qb   = (__bf16*)alloc(MT * Dm * 2);
    __bf16* kb   = (__bf16*)alloc(MT * Dm * 2);
    __bf16* vb   = (__bf16*)alloc(MT * Dm * 2);
    __bf16* yb   = (__bf16*)alloc(MT * Dm * 2);
    __bf16* kvT  = (__bf16*)alloc(64ull * 64 * 64 * 2);
    float*  sumk = (float*)alloc(64ull * 64 * 4);

    // staging: x -> bf16 row-major; weights -> bf16 transposed [N][K]
    cvt_f32_bf16<<<16384, 256, 0, stream>>>(x, xb, (int)(MT * Dm));
    cvt_transpose_f32_bf16<<<256, 256, 0, stream>>>(Wq, WqT, 1024, 1024);
    cvt_transpose_f32_bf16<<<256, 256, 0, stream>>>(Wk, WkT, 1024, 1024);
    cvt_transpose_f32_bf16<<<256, 256, 0, stream>>>(Wv, WvT, 1024, 1024);
    cvt_transpose_f32_bf16<<<256, 256, 0, stream>>>(Wp, WpT, 1024, 1024);

    dim3 grid(8, 128), blk(256);
    // q = elu(x Wq + bq)+1 ; k = elu(x Wk + bk)+1 ; v = x Wv + bv
    gemm_bf16<1, __bf16><<<grid, blk, 0, stream>>>(xb, WqT, bq, qb, 16384, 1024, 1024);
    gemm_bf16<1, __bf16><<<grid, blk, 0, stream>>>(xb, WkT, bk, kb, 16384, 1024, 1024);
    gemm_bf16<0, __bf16><<<grid, blk, 0, stream>>>(xb, WvT, bv, vb, 16384, 1024, 1024);

    // kv = k^T v per head, plus sum_k
    kv_sumk_kernel<<<64, 256, 0, stream>>>(kb, vb, kvT, sumk);

    // y = (q kv) / (q . sum_k), heads merged, bf16
    attn_num_kernel<<<dim3(32, 64), 256, 0, stream>>>(qb, kvT, sumk, yb);

    // out = y Wp + bp (f32 output)
    gemm_bf16<0, float><<<grid, blk, 0, stream>>>(yb, WpT, bp, out, 16384, 1024, 1024);
}
